// ResidualVectorQuantizer_48017734369464
// MI455X (gfx1250) — compile-verified, matrix operand order checked by reading, not run on hardware
//
#include <hip/hip_runtime.h>
#include <hip/hip_bf16.h>

// ---------------------------------------------------------------------------
// Residual Vector Quantization for MI455X (gfx1250, wave32, WMMA)
//   x:         (B=32, D=256, T=4096) f32
//   codebooks: (S=8,  K=1024, D=256) f32
//   out:       recon (B, D, T) f32  ++  indices (B, S, T) i32  (concatenated)
//
// recon accumulates in d_out; residual = x - recon recomputed per stage.
// Score GEMM r@W^T runs on the bf16 WMMA pipe with hi/lo f32 splitting
// (3 x v_wmma_f32_16x16x32_bf16 per effective f32 K=32 step) so the argmin
// stays f32-accurate. A-fragments are register-resident across the whole
// 1024-code sweep; only B fragments stream through LDS.
// ---------------------------------------------------------------------------

typedef __attribute__((ext_vector_type(16))) __bf16 bf16x16;
typedef __attribute__((ext_vector_type(8)))  float  f32x8;

#define B_DIM    32
#define D_DIM    256
#define T_DIM    4096
#define S_STAGES 8
#define K_CODES  1024

#define TM       64     // tokens per block (4 waves x 16 rows)
#define KC       32     // codes per chunk (2 x 16-wide N tiles per wave)
#define RS       272    // padded LDS row stride in bf16 elems (544 B, 16B-mult)
#define NTHREADS 128

union Frag {
    uint4   q[2];
    bf16x16 v;
};

__global__ void rvq_zero_recon(float* __restrict__ out, int n) {
    int i = blockIdx.x * blockDim.x + threadIdx.x;
    if (i < n) out[i] = 0.0f;
}

__global__ __launch_bounds__(NTHREADS)
void rvq_stage_kernel(const float* __restrict__ x,
                      const float* __restrict__ cb,     // (S,K,D)
                      float* __restrict__ out,          // recon (B,D,T)
                      int*   __restrict__ ind,          // (B,S,T)
                      int s)
{
    __shared__ __bf16 rhi[TM * RS];
    __shared__ __bf16 rlo[TM * RS];
    __shared__ __bf16 whi[KC * RS];
    __shared__ __bf16 wlo[KC * RS];
    __shared__ float  wnormC[KC];
    __shared__ int    idxLDS[TM];

    const int tid  = threadIdx.x;
    const int lane = tid & 31;
    const int half = lane >> 4;        // 0: lanes 0-15, 1: lanes 16-31
    const int lm   = lane & 15;
    const int m0   = (tid >> 5) * 16;  // wave's 16-token row band

    const int tileTok = blockIdx.x * TM;        // 64 | T, so one b per block
    const int b  = tileTok / T_DIM;
    const int t0 = tileTok % T_DIM;

    const float* xblk = x   + (size_t)b * D_DIM * T_DIM + t0;
    float*       oblk = out + (size_t)b * D_DIM * T_DIM + t0;
    const float* wstg = cb  + (size_t)s * K_CODES * D_DIM;

    // ---- Phase 1: residual tile -> bf16 hi/lo in LDS (coalesced over t) ----
    for (int i = tid; i < TM * D_DIM; i += NTHREADS) {
        int d = i >> 6;            // feature
        int j = i & 63;            // token within tile
        float r  = xblk[(size_t)d * T_DIM + j] - oblk[(size_t)d * T_DIM + j];
        __bf16 hi = (__bf16)r;
        __bf16 lo = (__bf16)(r - (float)hi);
        rhi[j * RS + d] = hi;
        rlo[j * RS + d] = lo;
    }
    __syncthreads();

    // ---- Phase 2: hoist this wave's A fragments into registers ----
    // 16-bit A layout: lanes 0-15 hold K=0..7,16..23; lanes 16-31 K=8..15,24..31
    bf16x16 ahi[8], alo[8];
    {
        const int arow = (m0 + lm) * RS;
#pragma unroll
        for (int fc = 0; fc < 8; ++fc) {
            const int fbase = fc * 32 + half * 8;
            Frag t;
            t.q[0] = *(const uint4*)&rhi[arow + fbase];
            t.q[1] = *(const uint4*)&rhi[arow + fbase + 16];
            ahi[fc] = t.v;
            t.q[0] = *(const uint4*)&rlo[arow + fbase];
            t.q[1] = *(const uint4*)&rlo[arow + fbase + 16];
            alo[fc] = t.v;
        }
    }

    float bestv[8];
    int   besti[8];
#pragma unroll
    for (int j = 0; j < 8; ++j) { bestv[j] = 3.4e38f; besti[j] = 0; }

    // ---- Phase 3: sweep codebook in chunks of 32 codes ----
    for (int kc = 0; kc < K_CODES / KC; ++kc) {
        __syncthreads();           // protect whi/wnormC reuse from prev chunk
        if (tid < KC) wnormC[tid] = 0.0f;
        __syncthreads();

        // cooperative load: 4 threads per code, 64 contiguous feats each
        {
            int code = tid >> 2;               // 0..31
            int f0   = (tid & 3) * 64;
            const float* wrow = wstg + (size_t)(kc * KC + code) * D_DIM + f0;
            float ss = 0.0f;
#pragma unroll 8
            for (int q = 0; q < 64; ++q) {
                float v = wrow[q];
                __bf16 hi = (__bf16)v;
                __bf16 lo = (__bf16)(v - (float)hi);
                whi[code * RS + f0 + q] = hi;
                wlo[code * RS + f0 + q] = lo;
                ss += v * v;
            }
            atomicAdd(&wnormC[code], ss);      // ds_add_f32
        }
        __syncthreads();

        // ---- WMMA: two 16x16 N-tiles per wave, K=256, hi/lo split ----
#pragma unroll
        for (int nt = 0; nt < 2; ++nt) {
            f32x8 c = {};
            const int brow = (nt * 16 + lm) * RS;
#pragma unroll
            for (int fc = 0; fc < 8; ++fc) {
                const int fbase = fc * 32 + half * 8;
                Frag bh, bl;
                bh.q[0] = *(const uint4*)&whi[brow + fbase];
                bh.q[1] = *(const uint4*)&whi[brow + fbase + 16];
                bl.q[0] = *(const uint4*)&wlo[brow + fbase];
                bl.q[1] = *(const uint4*)&wlo[brow + fbase + 16];

                c = __builtin_amdgcn_wmma_f32_16x16x32_bf16(
                        false, ahi[fc], false, bh.v, (short)0, c, false, false);
                c = __builtin_amdgcn_wmma_f32_16x16x32_bf16(
                        false, ahi[fc], false, bl.v, (short)0, c, false, false);
                c = __builtin_amdgcn_wmma_f32_16x16x32_bf16(
                        false, alo[fc], false, bh.v, (short)0, c, false, false);
            }

            // dist(n,k) = ||w_k||^2 - 2 <r_n, w_k>; merge into running argmin
            const float wn    = wnormC[nt * 16 + lm];
            const int   kthis = kc * KC + nt * 16 + lm;
#pragma unroll
            for (int j = 0; j < 8; ++j) {
                float dist = wn - 2.0f * c[j];
                if (dist < bestv[j]) { bestv[j] = dist; besti[j] = kthis; }
            }
        }
    }

    // ---- Phase 4: cross-lane argmin within each 16-lane half ----
#pragma unroll
    for (int j = 0; j < 8; ++j) {
        float v  = bestv[j];
        int   ii = besti[j];
#pragma unroll
        for (int m = 1; m <= 8; m <<= 1) {
            float ov = __shfl_xor(v, m, 32);
            int   oi = __shfl_xor(ii, m, 32);
            if (ov < v || (ov == v && oi < ii)) { v = ov; ii = oi; }
        }
        if (lm == 0) {
            int tokm = j + 8 * half;           // C layout: M = j + 8*(lane/16)
            idxLDS[m0 + tokm] = ii;
        }
    }
    __syncthreads();

    // ---- Phase 5: emit indices + accumulate recon (coalesced over t) ----
    if (tid < TM) {
        int tokg = tileTok + tid;
        int bb   = tokg >> 12;                 // / T_DIM
        int tt   = tokg & (T_DIM - 1);
        ind[(size_t)bb * S_STAGES * T_DIM + (size_t)s * T_DIM + tt] = idxLDS[tid];
    }

    for (int i = tid; i < TM * D_DIM; i += NTHREADS) {
        int d = i >> 6;
        int j = i & 63;
        int code = idxLDS[j];
        float wv = wstg[(size_t)code * D_DIM + d];
        oblk[(size_t)d * T_DIM + j] += wv;
    }
}

extern "C" void kernel_launch(void* const* d_in, const int* in_sizes, int n_in,
                              void* d_out, int out_size, void* d_ws, size_t ws_size,
                              hipStream_t stream) {
    (void)in_sizes; (void)n_in; (void)out_size; (void)d_ws; (void)ws_size;

    const float* x  = (const float*)d_in[0];   // (32, 256, 4096)
    const float* cb = (const float*)d_in[1];   // (8, 1024, 256)

    float* out = (float*)d_out;                          // recon (B,D,T)
    int*   ind = (int*)(out + (size_t)B_DIM * D_DIM * T_DIM);  // (B,S,T)

    const int reconN = B_DIM * D_DIM * T_DIM;            // 33,554,432
    rvq_zero_recon<<<(reconN + 255) / 256, 256, 0, stream>>>(out, reconN);

    const int nTokens = B_DIM * T_DIM;                   // 131,072
    dim3 grid(nTokens / TM);                             // 2048 blocks
    dim3 block(NTHREADS);                                // 4 waves (wave32)
    for (int s = 0; s < S_STAGES; ++s) {
        rvq_stage_kernel<<<grid, block, 0, stream>>>(x, cb, out, ind, s);
    }
}
